// ConditionedEpsilonTheta_85246510891542
// MI455X (gfx1250) — matrix operand order repaired and checked by reading, not compiled, not run on hardware
//
#include <hip/hip_runtime.h>
#include <hip/hip_bf16.h>

// ---------------------------------------------------------------------------
// Problem dims (fixed by the reference)
// ---------------------------------------------------------------------------
#define B_   16
#define C_   256
#define H_   1024
#define S_   1024
#define S1_  1025
#define E_   512
#define NH_  8
#define HD_  64
#define DSZ_ 64
#define DD_  512

typedef __attribute__((ext_vector_type(16))) __bf16 v16bf;
typedef __attribute__((ext_vector_type(8)))  __bf16 v8bf;
typedef __attribute__((ext_vector_type(4)))  __bf16 v4bf;
typedef __attribute__((ext_vector_type(8)))  float  v8f;

static __device__ __forceinline__ v8f wmma_bf16(v16bf a, v16bf b, v8f c) {
  return __builtin_amdgcn_wmma_f32_16x16x32_bf16(false, a, false, b, (short)0, c,
                                                 false, false);
}

// LDS byte offset of a __shared__ object (generic addr low 32 bits = LDS addr)
static __device__ __forceinline__ unsigned lds_off(const void* p) {
  return (unsigned)(unsigned long long)p;
}

// CDNA5 async global->LDS copy, 16 bytes per lane, tracked by ASYNCcnt
static __device__ __forceinline__ void async_copy_b128(unsigned lds, const void* g) {
  asm volatile("global_load_async_to_lds_b128 %0, %1, off"
               :: "v"(lds), "v"((unsigned long long)g) : "memory");
}
static __device__ __forceinline__ void wait_async0() {
  asm volatile("s_wait_asynccnt 0x0" ::: "memory");
}

// ---------------------------------------------------------------------------
// fp32 -> bf16 bulk conversion (4 elems / thread)
// ---------------------------------------------------------------------------
__global__ __launch_bounds__(256) void cvt_bf16_kernel(
    const float* __restrict__ src, __bf16* __restrict__ dst, int n4) {
  int i = blockIdx.x * 256 + threadIdx.x;
  if (i < n4) {
    float4 v = reinterpret_cast<const float4*>(src)[i];
    v4bf o;
    o[0] = (__bf16)v.x; o[1] = (__bf16)v.y; o[2] = (__bf16)v.z; o[3] = (__bf16)v.w;
    reinterpret_cast<v4bf*>(dst)[i] = o;
  }
}

// ---------------------------------------------------------------------------
// x[B,C,H] -> xT_bf16[B*H, C]   (32x32 LDS tile transpose, coalesced both ways)
// ---------------------------------------------------------------------------
__global__ __launch_bounds__(256) void xpose_kernel(
    const float* __restrict__ x, __bf16* __restrict__ xT) {
  const int blk = blockIdx.x;
  const int b  = blk >> 8;
  const int cy = (blk >> 5) & 7;   // C tile (8)
  const int hx = blk & 31;         // H tile (32)
  __shared__ float t[32][33];
  const int col = threadIdx.x & 31;
  const int r0  = (threadIdx.x >> 5) * 4;
  for (int i = 0; i < 4; ++i) {
    int cl = r0 + i;
    t[cl][col] = x[((long)(b * C_ + cy * 32 + cl)) * H_ + hx * 32 + col];
  }
  __syncthreads();
  for (int i = 0; i < 4; ++i) {
    int hl = r0 + i;
    xT[((long)(b * H_ + hx * 32 + hl)) * C_ + cy * 32 + col] = (__bf16)t[col][hl];
  }
}

// ---------------------------------------------------------------------------
// K1: static_token (B,E) fp32, FiLM gate (B,C) fp32; static row of cond (bf16)
// ---------------------------------------------------------------------------
__global__ __launch_bounds__(512) void static_film_kernel(
    const float* __restrict__ stat, const float* __restrict__ stat_W,
    const float* __restrict__ stat_b,
    const float* __restrict__ film_W1, const float* __restrict__ film_b1,
    const float* __restrict__ film_W2, const float* __restrict__ film_b2,
    float* __restrict__ ws_film, __bf16* __restrict__ cond_bf) {
  const int b = blockIdx.x;
  const int t = threadIdx.x;
  __shared__ float st[E_];
  __shared__ float h1[E_];
  {
    float s = stat_b[t];
    for (int j = 0; j < DSZ_; ++j) s += stat[b * DSZ_ + j] * stat_W[t * DSZ_ + j];
    st[t] = s;
    cond_bf[((long)(b * S1_ + S_)) * E_ + t] = (__bf16)s;  // static token row
  }
  __syncthreads();
  {
    float s = film_b1[t];
    for (int j = 0; j < E_; ++j) s += st[j] * film_W1[t * E_ + j];
    h1[t] = (s >= 0.f) ? s : 0.1f * s;  // leaky
  }
  __syncthreads();
  if (t < C_) {
    float s = film_b2[t];
    for (int j = 0; j < E_; ++j) s += h1[j] * film_W2[t * E_ + j];
    ws_film[b * C_ + t] = tanhf(s);
  }
}

// ---------------------------------------------------------------------------
// K2: Wcomb = (Wq @ q_W) * 0.125 (bf16);  bcomb = (bq + Wq·q_b) * 0.125 (fp32)
// 1/8 = 1/sqrt(HD) folded in so attention consumes pre-scaled q.
// ---------------------------------------------------------------------------
__global__ __launch_bounds__(256) void wcomb_kernel(
    const float* __restrict__ in_proj_W, const float* __restrict__ in_proj_b,
    const float* __restrict__ q_W, const float* __restrict__ q_b,
    __bf16* __restrict__ Wcomb_bf, float* __restrict__ bcomb) {
  const int e = blockIdx.x;   // 0..511
  const int c = threadIdx.x;  // 0..255
  float s = 0.f;
  for (int j = 0; j < E_; ++j) s += in_proj_W[e * E_ + j] * q_W[j * C_ + c];
  Wcomb_bf[e * C_ + c] = (__bf16)(s * 0.125f);
  if (c == 0) {
    float bsum = in_proj_b[e];
    for (int j = 0; j < E_; ++j) bsum += in_proj_W[e * E_ + j] * q_b[j];
    bcomb[e] = bsum * 0.125f;
  }
}

// ---------------------------------------------------------------------------
// bf16 WMMA GEMM:  D[m,n] = sum_k A[m,k] * W[n,k] + bias[n]
// A,W bf16 row-major; D fp32 or bf16 (flag); out row = m + (m/outSeg)*outPad.
// 256 threads (8 waves), tile 128x128, BK=32, async global->LDS tile moves.
// OOB A rows are CLAMPED to M-1 (branch-free loads); store guard discards them.
// ---------------------------------------------------------------------------
__global__ __launch_bounds__(256) void gemm_bf16_kernel(
    const __bf16* __restrict__ A, const __bf16* __restrict__ W,
    const float* __restrict__ bias, void* __restrict__ Dvoid,
    int M, int N, int K, int outSeg, int outPad, int out_bf16) {
  __shared__ __attribute__((aligned(64))) __bf16 As[128 * 32];
  __shared__ __attribute__((aligned(64))) __bf16 Bs[128 * 32];
  const int tid = threadIdx.x;
  const int n0 = blockIdx.x * 128;
  const int m0 = blockIdx.y * 128;
  const int wv = tid >> 5, lane = tid & 31;
  const int lh = lane >> 4, ll = lane & 15;
  const int wm = wv >> 2, wn = wv & 3;
  const unsigned asb = lds_off(&As[0]);
  const unsigned bsb = lds_off(&Bs[0]);

  v8f acc[4][2] = {};

  for (int k0 = 0; k0 < K; k0 += 32) {
    // A tile: 128x32 bf16 = 512 x 16B chunks, 2 per thread, branch-free async
    for (int i = 0; i < 2; ++i) {
      int chunk = i * 256 + tid;
      int row = chunk >> 2, c8 = chunk & 3;
      int m = m0 + row;
      m = (m < M) ? m : (M - 1);  // clamp: duplicated rows discarded at store
      const __bf16* g = A + (long)m * K + (k0 + c8 * 8);
      async_copy_b128(asb + (unsigned)(row * 32 + c8 * 8) * 2, g);
      if (k0 + 32 < K) __builtin_prefetch(g + 32, 0, 1);  // next K-step chunk
    }
    // W tile (N tiles always full)
    for (int i = 0; i < 2; ++i) {
      int chunk = i * 256 + tid;
      int row = chunk >> 2, c8 = chunk & 3;
      async_copy_b128(bsb + (unsigned)(row * 32 + c8 * 8) * 2,
                      W + (long)(n0 + row) * K + (k0 + c8 * 8));
    }
    wait_async0();
    __syncthreads();

    // fragments (CDNA5 wave32 layouts), vector LDS loads
    v16bf af[4];
    for (int i = 0; i < 4; ++i) {
      int ml = wm * 64 + i * 16 + ll;
      v8bf lo = *reinterpret_cast<const v8bf*>(&As[ml * 32 + lh * 16]);
      v8bf hi = *reinterpret_cast<const v8bf*>(&As[ml * 32 + lh * 16 + 8]);
      af[i] = __builtin_shufflevector(lo, hi, 0, 1, 2, 3, 4, 5, 6, 7,
                                      8, 9, 10, 11, 12, 13, 14, 15);
    }
    v16bf bfr[2];
    for (int j = 0; j < 2; ++j) {
      int nl = wn * 32 + j * 16 + ll;
      bfr[j] = *reinterpret_cast<const v16bf*>(&Bs[nl * 32 + lh * 16]);
    }
    for (int i = 0; i < 4; ++i)
      for (int j = 0; j < 2; ++j)
        acc[i][j] = wmma_bf16(af[i], bfr[j], acc[i][j]);
    __syncthreads();
  }

  for (int i = 0; i < 4; ++i)
    for (int j = 0; j < 2; ++j) {
      int n = n0 + wn * 32 + j * 16 + ll;
      float bv = bias[n];
      for (int r = 0; r < 8; ++r) {
        int m = m0 + wm * 64 + i * 16 + lh * 8 + r;
        if (m < M) {
          long orow = (long)m + (long)(m / outSeg) * outPad;
          float val = acc[i][j][r] + bv;
          if (out_bf16)
            ((__bf16*)Dvoid)[orow * (long)N + n] = (__bf16)val;
          else
            ((float*)Dvoid)[orow * (long)N + n] = val;
        }
      }
    }
}

// ---------------------------------------------------------------------------
// Flash attention: grid = B*NH*(H/64), 128 threads (4 waves).
// q pre-scaled bf16; K/V chunks async-copied to LDS (keys clamped to the
// static token row -> masked to weight 0, so no divergent zero-fill needed);
// online softmax with cross-lane reductions; QK^T and P·V via bf16 WMMA.
// ---------------------------------------------------------------------------
__global__ __launch_bounds__(128) void attn_kernel(
    const __bf16* __restrict__ qbf,        // [B*H, E]
    const __bf16* __restrict__ kvbf,       // [B*S1, 2E] (k | v)
    const float* __restrict__ bias_table,  // [4095]
    __bf16* __restrict__ ctx_bf) {         // [B*H, E]
  const int qt = blockIdx.x & 15;
  const int hdi = (blockIdx.x >> 4) & 7;  // head
  const int b  = blockIdx.x >> 7;
  const int tid = threadIdx.x, w = tid >> 5, lane = tid & 31;
  const int lh = lane >> 4, ll = lane & 15;
  const int qb = qt * 64;
  const int qbase = qb + w * 16;

  __shared__ __attribute__((aligned(64))) __bf16 Kt[64][64];
  __shared__ __attribute__((aligned(64))) __bf16 Vt[64][64];
  __shared__ __attribute__((aligned(64))) __bf16 Ps[4][16][64];
  const unsigned ktb = lds_off(&Kt[0][0]);
  const unsigned vtb = lds_off(&Vt[0][0]);

  // Q A-fragments (two K-steps over HD=64), direct bf16 vector loads
  const __bf16* qrow = qbf + ((long)(b * H_ + qbase + ll)) * E_ + hdi * HD_;
  v16bf aq[2];
  for (int ks = 0; ks < 2; ++ks) {
    v8bf lo = *reinterpret_cast<const v8bf*>(qrow + ks * 32 + lh * 16);
    v8bf hi = *reinterpret_cast<const v8bf*>(qrow + ks * 32 + lh * 16 + 8);
    aq[ks] = __builtin_shufflevector(lo, hi, 0, 1, 2, 3, 4, 5, 6, 7,
                                     8, 9, 10, 11, 12, 13, 14, 15);
  }

  float mrow[8], lrow[8];
  v8f o[4] = {};
  for (int r = 0; r < 8; ++r) { mrow[r] = -3.0e38f; lrow[r] = 0.f; }

  const int lastc = qb >> 6;  // causal chunks 0..lastc, then static chunk 16
  for (int cc = 0; cc <= lastc + 1; ++cc) {
    const int c = (cc <= lastc) ? cc : 16;
    const int key0 = c * 64;

    // K & V chunk (each 64x64 bf16 = 512 chunks): 1024 / 128 thr = 8 each.
    // Branch-free: clamp OOB keys to the static-token row (weight 0 later).
    for (int i = 0; i < 8; ++i) {
      int idx = i * 128 + tid;
      int sel = idx >> 9;          // 0 = K, 1 = V
      int c2 = idx & 511;
      int row = c2 >> 3, c8 = c2 & 7;
      int key = key0 + row;
      key = (key < S1_) ? key : S_;  // clamp (always in-bounds)
      const __bf16* g = kvbf + ((long)(b * S1_ + key)) * (2 * E_) +
                        hdi * HD_ + sel * E_ + c8 * 8;
      async_copy_b128((sel ? vtb : ktb) + (unsigned)(row * 64 + c8 * 8) * 2, g);
    }
    wait_async0();
    __syncthreads();

    // S = Q K^T (16x64): 4 N-tiles x 2 K-steps
    v8f sc[4];
    for (int j = 0; j < 4; ++j) {
      v8f a0 = {};
      for (int ks = 0; ks < 2; ++ks) {
        v16bf bk = *reinterpret_cast<const v16bf*>(&Kt[j * 16 + ll][ks * 32 + lh * 16]);
        a0 = wmma_bf16(aq[ks], bk, a0);
      }
      sc[j] = a0;
    }

    // fused relative-position bias + causal mask
    for (int j = 0; j < 4; ++j) {
      int key = key0 + j * 16 + ll;
      for (int r = 0; r < 8; ++r) {
        int qrw = qbase + lh * 8 + r;
        float s = sc[j][r];
        if (key >= S1_) {
          s = -3.0e30f;                         // padded keys -> weight 0
        } else {
          s += bias_table[qrw - key + 2047];
          if (key < S_ && key > qrw) s += -1.0e9f;  // causal NEG
        }
        sc[j][r] = s;
      }
    }

    // online softmax (rows across 16 lanes of each half-wave)
    for (int r = 0; r < 8; ++r) {
      float mx = fmaxf(fmaxf(sc[0][r], sc[1][r]), fmaxf(sc[2][r], sc[3][r]));
      for (int m = 1; m < 16; m <<= 1) mx = fmaxf(mx, __shfl_xor(mx, m, 32));
      float mnew = fmaxf(mrow[r], mx);
      float alpha = __expf(mrow[r] - mnew);
      float psum = 0.f;
      for (int j = 0; j < 4; ++j) {
        float p = __expf(sc[j][r] - mnew);
        sc[j][r] = p;
        psum += p;
      }
      for (int m = 1; m < 16; m <<= 1) psum += __shfl_xor(psum, m, 32);
      lrow[r] = lrow[r] * alpha + psum;
      mrow[r] = mnew;
      for (int j = 0; j < 4; ++j) o[j][r] *= alpha;
    }

    // P: C-layout -> A-layout via per-wave LDS scratch
    for (int j = 0; j < 4; ++j)
      for (int r = 0; r < 8; ++r)
        Ps[w][lh * 8 + r][j * 16 + ll] = (__bf16)sc[j][r];
    __syncthreads();

    v16bf pa[2];
    for (int ks = 0; ks < 2; ++ks) {
      v8bf lo = *reinterpret_cast<const v8bf*>(&Ps[w][ll][ks * 32 + lh * 16]);
      v8bf hi = *reinterpret_cast<const v8bf*>(&Ps[w][ll][ks * 32 + lh * 16 + 8]);
      pa[ks] = __builtin_shufflevector(lo, hi, 0, 1, 2, 3, 4, 5, 6, 7,
                                       8, 9, 10, 11, 12, 13, 14, 15);
    }
    // O += P V  (4 hd-tiles x 2 key K-steps); V columns gathered from LDS
    for (int j = 0; j < 4; ++j) {
      for (int ks = 0; ks < 2; ++ks) {
        v16bf vb;
        for (int e = 0; e < 16; ++e)
          vb[e] = Vt[ks * 32 + lh * 16 + e][j * 16 + ll];
        o[j] = wmma_bf16(pa[ks], vb, o[j]);
      }
    }
    __syncthreads();
  }

  // normalize + store ctx (bf16)
  for (int j = 0; j < 4; ++j)
    for (int r = 0; r < 8; ++r) {
      int h = qbase + lh * 8 + r;
      float val = o[j][r] / lrow[r];
      ctx_bf[((long)(b * H_ + h)) * E_ + hdi * HD_ + j * 16 + ll] = (__bf16)val;
    }
}

// ---------------------------------------------------------------------------
// Final: out[b,c,h] = (x[b,c,h] + res[b,h,c]) * (1 + film[b,c])
// ---------------------------------------------------------------------------
__global__ __launch_bounds__(256) void final_kernel(
    const float* __restrict__ x, const float* __restrict__ res,
    const float* __restrict__ film, float* __restrict__ out) {
  const int b = blockIdx.x / C_;
  const int c = blockIdx.x % C_;
  const float scale = 1.f + film[b * C_ + c];
  for (int i = 0; i < 4; ++i) {
    int h = threadIdx.x + i * 256;
    long xi = ((long)(b * C_ + c)) * H_ + h;
    out[xi] = (x[xi] + res[((long)(b * H_ + h)) * C_ + c]) * scale;
  }
}

// ---------------------------------------------------------------------------
extern "C" void kernel_launch(void* const* d_in, const int* in_sizes, int n_in,
                              void* d_out, int out_size, void* d_ws, size_t ws_size,
                              hipStream_t stream) {
  const float* x         = (const float*)d_in[0];
  const float* dyn       = (const float*)d_in[1];
  const float* stat      = (const float*)d_in[2];
  const float* dyn_W     = (const float*)d_in[3];
  const float* dyn_b     = (const float*)d_in[4];
  const float* stat_W    = (const float*)d_in[5];
  const float* stat_b    = (const float*)d_in[6];
  const float* film_W1   = (const float*)d_in[7];
  const float* film_b1   = (const float*)d_in[8];
  const float* film_W2   = (const float*)d_in[9];
  const float* film_b2   = (const float*)d_in[10];
  const float* q_W       = (const float*)d_in[11];
  const float* q_b       = (const float*)d_in[12];
  const float* in_proj_W = (const float*)d_in[13];
  const float* in_proj_b = (const float*)d_in[14];
  const float* out_W     = (const float*)d_in[15];
  const float* out_b     = (const float*)d_in[16];
  const float* ctx_W     = (const float*)d_in[17];
  const float* ctx_b     = (const float*)d_in[18];
  const float* bias_table= (const float*)d_in[19];

  // ---- workspace carve (bytes, 16B aligned) ----
  char* wsb = (char*)d_ws;
  size_t cur = 0;
  auto carve = [&](size_t bytes) {
    size_t o = cur;
    cur += (bytes + 15) & ~(size_t)15;
    return o;
  };
  __bf16* dyn_bf    = (__bf16*)(wsb + carve((size_t)B_ * S_ * DD_ * 2));
  __bf16* dynW_bf   = (__bf16*)(wsb + carve((size_t)E_ * DD_ * 2));
  __bf16* kvW_bf    = (__bf16*)(wsb + carve((size_t)2 * E_ * E_ * 2));
  __bf16* outW_bf   = (__bf16*)(wsb + carve((size_t)E_ * E_ * 2));
  __bf16* ctxW_bf   = (__bf16*)(wsb + carve((size_t)C_ * E_ * 2));
  __bf16* wcomb_bf  = (__bf16*)(wsb + carve((size_t)E_ * C_ * 2));
  __bf16* xT_bf     = (__bf16*)(wsb + carve((size_t)B_ * H_ * C_ * 2));
  __bf16* cond_bf   = (__bf16*)(wsb + carve((size_t)B_ * S1_ * E_ * 2));
  __bf16* q_bf      = (__bf16*)(wsb + carve((size_t)B_ * H_ * E_ * 2));
  __bf16* kv_bf     = (__bf16*)(wsb + carve((size_t)B_ * S1_ * 2 * E_ * 2));
  __bf16* ctx_bf    = (__bf16*)(wsb + carve((size_t)B_ * H_ * E_ * 2));
  __bf16* attn_bf   = (__bf16*)(wsb + carve((size_t)B_ * H_ * E_ * 2));
  float* ws_film = (float*)(wsb + carve((size_t)B_ * C_ * 4));
  float* bcomb   = (float*)(wsb + carve((size_t)E_ * 4));
  float* res     = (float*)(wsb + carve((size_t)B_ * H_ * C_ * 4));

  auto cvt = [&](const float* s, __bf16* d, long n) {
    int n4 = (int)(n / 4);
    cvt_bf16_kernel<<<(n4 + 255) / 256, 256, 0, stream>>>(s, d, n4);
  };

  // 0) precision conversions (one cheap pass each)
  cvt(dyn, dyn_bf, (long)B_ * S_ * DD_);
  cvt(dyn_W, dynW_bf, (long)E_ * DD_);
  cvt(in_proj_W + (size_t)E_ * E_, kvW_bf, (long)2 * E_ * E_);
  cvt(out_W, outW_bf, (long)E_ * E_);
  cvt(ctx_W, ctxW_bf, (long)C_ * E_);
  xpose_kernel<<<B_ * (C_ / 32) * (H_ / 32), 256, 0, stream>>>(x, xT_bf);

  // 1) static token + FiLM (fills cond static rows)
  static_film_kernel<<<B_, 512, 0, stream>>>(
      stat, stat_W, stat_b, film_W1, film_b1, film_W2, film_b2, ws_film, cond_bf);

  // 2) fused, pre-scaled query weight
  wcomb_kernel<<<E_, 256, 0, stream>>>(in_proj_W, in_proj_b, q_W, q_b,
                                       wcomb_bf, bcomb);

  // 3) dyn_tokens -> cond (bf16, row-gapped into [B,S1,E])
  gemm_bf16_kernel<<<dim3(E_ / 128, (B_ * S_) / 128), 256, 0, stream>>>(
      dyn_bf, dynW_bf, dyn_b, cond_bf, B_ * S_, E_, DD_, S_, 1, 1);

  // 4) q = xT @ Wcomb^T (pre-scaled)
  gemm_bf16_kernel<<<dim3(E_ / 128, (B_ * H_) / 128), 256, 0, stream>>>(
      xT_bf, wcomb_bf, bcomb, q_bf, B_ * H_, E_, C_, B_ * H_, 0, 1);

  // 5) fused k|v projection
  gemm_bf16_kernel<<<dim3((2 * E_) / 128, (B_ * S1_ + 127) / 128), 256, 0, stream>>>(
      cond_bf, kvW_bf, in_proj_b + E_, kv_bf, B_ * S1_, 2 * E_, E_,
      B_ * S1_, 0, 1);

  // 6) flash attention
  attn_kernel<<<B_ * NH_ * (H_ / 64), 128, 0, stream>>>(
      q_bf, kv_bf, bias_table, ctx_bf);

  // 7) output projection
  gemm_bf16_kernel<<<dim3(E_ / 128, (B_ * H_) / 128), 256, 0, stream>>>(
      ctx_bf, outW_bf, out_b, attn_bf, B_ * H_, E_, E_, B_ * H_, 0, 1);

  // 8) conditioning residual (fp32 out)
  gemm_bf16_kernel<<<dim3(C_ / 128, (B_ * H_) / 128), 256, 0, stream>>>(
      attn_bf, ctxW_bf, ctx_b, res, B_ * H_, C_, E_, B_ * H_, 0, 0);

  // 9) (x + residual) * (1 + film)
  final_kernel<<<B_ * C_, 256, 0, stream>>>(x, res, ws_film, (float*)d_out);
}
